// QuantizeEMAResetL2_12421045420158
// MI455X (gfx1250) — matrix operand order repaired and checked by reading, not compiled, not run on hardware
//
#include <hip/hip_runtime.h>
#include <hip/hip_bf16.h>

typedef __attribute__((ext_vector_type(16))) _Float16 v16h;
typedef __attribute__((ext_vector_type(8)))  float    v8f;

#define N_BATCH 32
#define D_DIM   512
#define T_DIM   2048
#define NB_CODE 512
#define NT      (N_BATCH * T_DIM)          // 65536 rows
#define WAVES_PER_BLOCK 8
#define ROWS_PER_WAVE   16
#define GEMM_BLOCKS (NT / (WAVES_PER_BLOCK * ROWS_PER_WAVE))   // 512
#define N_PARTIALS  (GEMM_BLOCKS * WAVES_PER_BLOCK)            // 4096

// ---------------------------------------------------------------------------
// K0: codebook f32 -> f16, per-code squared norm cc[n], zero histogram
// ---------------------------------------------------------------------------
__global__ __launch_bounds__(128) void k_prep(const float* __restrict__ cb,
                                              _Float16* __restrict__ cb16,
                                              float* __restrict__ cc,
                                              int* __restrict__ counts) {
    const int code = blockIdx.x;                 // 512 blocks, one code each
    __shared__ float red[128];
    float s = 0.0f;
    for (int d = threadIdx.x; d < D_DIM; d += 128) {
        float v = cb[(size_t)code * D_DIM + d];
        cb16[(size_t)code * D_DIM + d] = (_Float16)v;
        s += v * v;
    }
    red[threadIdx.x] = s;
    __syncthreads();
    for (int o = 64; o > 0; o >>= 1) {
        if (threadIdx.x < o) red[threadIdx.x] += red[threadIdx.x + o];
        __syncthreads();
    }
    if (threadIdx.x == 0) {
        cc[code] = red[0];
        counts[code] = 0;
    }
}

// ---------------------------------------------------------------------------
// K1: fused normalize + GEMM (WMMA f16 -> f32) + row argmin + histogram
//     + per-wave commitment-loss partials.
// Each wave owns 16 rows; full K=512 of A lives in 16 v16h fragments (regs).
// ---------------------------------------------------------------------------
__global__ __launch_bounds__(256) void k_gemm_argmin(
        const float* __restrict__ x,          // (32, 512, 2048)
        const _Float16* __restrict__ cb16,    // (512, 512) f16
        const float* __restrict__ cc,         // (512) ||cb||^2
        int* __restrict__ code_idx,           // (65536)
        int* __restrict__ counts,             // (512)
        float* __restrict__ commit_partials)  // (4096)
{
    const int wave = threadIdx.x >> 5;
    const int lane = threadIdx.x & 31;
    const int m    = lane & 15;               // row within wave tile / column id
    const int hi   = lane >> 4;               // half-wave select

    const long row_base = ((long)blockIdx.x * WAVES_PER_BLOCK + wave) * ROWS_PER_WAVE;
    const int  n  = (int)(row_base >> 11);    // batch index  (row = n*T + t)
    const int  t0 = (int)(row_base & (T_DIM - 1));
    const float* xn = x + (size_t)n * D_DIM * T_DIM;

    // ---- load A tile (raw x, f32) -> f16 fragments; accumulate sum(x^2) ----
    // A 16x32 f16 layout (matches B): lane<16 row=m holds K {0..15},
    // lane>=16 row=m holds K {16..31}.  k_local = h + 16*hi.
    v16h a[16];
    float sumsq = 0.0f;
#pragma unroll
    for (int kc = 0; kc < 16; ++kc) {
#pragma unroll
        for (int h = 0; h < 16; ++h) {
            const int kl = h + (hi << 4);
            const int k  = (kc << 5) + kl;
            float v = xn[(size_t)k * T_DIM + t0 + m];
            sumsq += v * v;
            a[kc][h] = (_Float16)v;
        }
    }
    sumsq += __shfl_xor(sumsq, 16, 32);                       // full-row sum
    const float inv = 1.0f / fmaxf(sqrtf(sumsq), 1e-12f);     // F.normalize clamp
    const float xx  = sumsq * inv * inv;                      // ||xf||^2

    // Accumulator VGPR v holds row (v + 8*hi); fetch that row's inv.
    float invrow[8];
#pragma unroll
    for (int v = 0; v < 8; ++v)
        invrow[v] = __shfl(inv, v + (hi << 3), 32);

    float best_val[8];
    int   best_idx[8];
#pragma unroll
    for (int v = 0; v < 8; ++v) { best_val[v] = 3.0e38f; best_idx[v] = 0; }

    // ---- sweep 512 codes in 16-wide chunks ----
    for (int j0 = 0; j0 < NB_CODE; j0 += 16) {
        const float ccj = cc[j0 + m];                         // this lane's column
        v8f acc = {0.f, 0.f, 0.f, 0.f, 0.f, 0.f, 0.f, 0.f};
        // B 32x16 f16 layout: lane = column (j0+m); lane<16 K 0..15, lane>=16 K 16..31
        const _Float16* bp = cb16 + (size_t)(j0 + m) * D_DIM + (hi << 4);
#pragma unroll
        for (int kc = 0; kc < 16; ++kc) {
            v16h b = *(const v16h*)(bp + (kc << 5));
            acc = __builtin_amdgcn_wmma_f32_16x16x32_f16(
                      false, a[kc], false, b, (short)0, acc, false, false);
        }
#pragma unroll
        for (int v = 0; v < 8; ++v) {
            // dist - xx  (xx constant per row; restored for the loss below)
            const float d = ccj - 2.0f * invrow[v] * acc[v];
            if (d < best_val[v]) { best_val[v] = d; best_idx[v] = j0 + m; }
        }
    }

    // ---- argmin across the 16 columns held by the 16-lane group ----
#pragma unroll
    for (int v = 0; v < 8; ++v) {
#pragma unroll
        for (int off = 1; off < 16; off <<= 1) {
            float ov = __shfl_xor(best_val[v], off, 32);
            int   oi = __shfl_xor(best_idx[v], off, 32);
            if (ov < best_val[v] || (ov == best_val[v] && oi < best_idx[v])) {
                best_val[v] = ov; best_idx[v] = oi;
            }
        }
    }

    // ---- commitment partial: sum over 16 rows of (xx + min_dist_proxy) ----
    float s0 = 0.0f;
#pragma unroll
    for (int v = 0; v < 8; ++v) s0 += best_val[v];
    s0 += __shfl_xor(s0, 16, 32);                 // both half-rows sets
    float tx = xx;
#pragma unroll
    for (int off = 1; off < 16; off <<= 1) tx += __shfl_xor(tx, off, 32);
    if (lane == 0)
        commit_partials[blockIdx.x * WAVES_PER_BLOCK + wave] = s0 + tx;

    // ---- emit indices + histogram (lane 0 -> rows 0..7, lane 16 -> 8..15) ----
    if (m == 0) {
#pragma unroll
        for (int v = 0; v < 8; ++v) {
            const long r = row_base + (hi << 3) + v;
            code_idx[r] = best_idx[v];
            atomicAdd(&counts[best_idx[v]], 1);
        }
    }
}

// ---------------------------------------------------------------------------
// K2: out[n,d,t] = codebook[idx(n,t), d]   (STE output == x_d numerically)
// ---------------------------------------------------------------------------
__global__ __launch_bounds__(256) void k_gather(const float* __restrict__ cb,
                                                const int* __restrict__ code_idx,
                                                float* __restrict__ out) {
    const long r = (long)blockIdx.x * 256 + threadIdx.x;     // row id in [0,65536)
    const int  n = (int)(r >> 11);
    const int  t = (int)(r & (T_DIM - 1));
    const int  idx = code_idx[r];
    const float* c = cb + (size_t)idx * D_DIM;
    float* o = out + (size_t)n * D_DIM * T_DIM + t;
#pragma unroll 8
    for (int d = 0; d < D_DIM; ++d)
        o[(size_t)d * T_DIM] = c[d];
}

// ---------------------------------------------------------------------------
// K3: finalize commit loss + perplexity (single block, fixed-order = determ.)
// ---------------------------------------------------------------------------
__global__ __launch_bounds__(256) void k_final(const float* __restrict__ partials,
                                               const int* __restrict__ counts,
                                               float* __restrict__ scalars) {
    __shared__ float red[256];
    // commit loss
    float s = 0.0f;
    for (int i = threadIdx.x * 16; i < threadIdx.x * 16 + 16; ++i) s += partials[i];
    red[threadIdx.x] = s;
    __syncthreads();
    for (int o = 128; o > 0; o >>= 1) {
        if (threadIdx.x < o) red[threadIdx.x] += red[threadIdx.x + o];
        __syncthreads();
    }
    if (threadIdx.x == 0)
        scalars[0] = red[0] / ((float)NT * (float)D_DIM);
    __syncthreads();
    // perplexity
    float p = 0.0f;
    for (int c = threadIdx.x; c < NB_CODE; c += 256) {
        float pr = (float)counts[c] * (1.0f / (float)NT);
        p += pr * logf(pr + 1e-7f);
    }
    red[threadIdx.x] = p;
    __syncthreads();
    for (int o = 128; o > 0; o >>= 1) {
        if (threadIdx.x < o) red[threadIdx.x] += red[threadIdx.x + o];
        __syncthreads();
    }
    if (threadIdx.x == 0)
        scalars[1] = expf(-red[0]);
}

// ---------------------------------------------------------------------------
extern "C" void kernel_launch(void* const* d_in, const int* in_sizes, int n_in,
                              void* d_out, int out_size, void* d_ws, size_t ws_size,
                              hipStream_t stream) {
    const float* x  = (const float*)d_in[0];   // (32, 512, 2048) f32
    const float* cb = (const float*)d_in[1];   // (512, 512) f32, pre-normalized
    float* out = (float*)d_out;                // 32*512*2048 + 2 floats

    char* w = (char*)d_ws;
    _Float16* cb16     = (_Float16*)(w);                  // 512*512*2   = 524288 B
    float*    cc       = (float*)(w + 524288);            // 512*4       = 2048 B
    int*      counts   = (int*)(w + 526336);              // 512*4       = 2048 B
    int*      code_idx = (int*)(w + 528384);              // 65536*4     = 262144 B
    float*    partials = (float*)(w + 790528);            // 4096*4      = 16384 B

    k_prep       <<<NB_CODE,     128, 0, stream>>>(cb, cb16, cc, counts);
    k_gemm_argmin<<<GEMM_BLOCKS, 256, 0, stream>>>(x, cb16, cc, code_idx, counts, partials);
    k_gather     <<<NT / 256,    256, 0, stream>>>(cb, code_idx, out);
    k_final      <<<1,           256, 0, stream>>>(partials, counts,
                                                   out + (size_t)NT * D_DIM);
}
